// GAT_81836306858559
// MI455X (gfx1250) — matrix operand order repaired and checked by reading, not compiled, not run on hardware
//
#include <hip/hip_runtime.h>

#define HID 64
#define NH  4
#define CH  16

typedef __attribute__((ext_vector_type(16))) _Float16 v16h;
typedef __attribute__((ext_vector_type(8)))  float    v8f;

__device__ __forceinline__ unsigned f2ord(float f) {
    unsigned u = __float_as_uint(f);
    return (u & 0x80000000u) ? ~u : (u | 0x80000000u);
}
__device__ __forceinline__ float ord2f(unsigned u) {
    return (u & 0x80000000u) ? __uint_as_float(u & 0x7fffffffu)
                             : __uint_as_float(~u);
}
__device__ __forceinline__ float lrelu(float x, float s) { return x > 0.f ? x : s * x; }

// ---------------------------------------------------------------------------
// Weight prep: f32 [FIN,64] -> f16 fragments in WMMA B layout:
//   frag[((s*4 + t)*32 + lane)*16 + i] = w[(s*32 + kb + (i&7) + (i>>3)*16)*64 + t*16 + (lane&15)]
//   kb = (lane<16) ? 0 : 8   (B elem i <-> K = kb+(i&7)+(i>>3)*16, col = lane&15)
// One thread emits one lane-fragment (16 contiguous halves = 32 B).
// ---------------------------------------------------------------------------
template <int FIN>
__global__ void prep_wfrag(const float* __restrict__ w, _Float16* __restrict__ frag)
{
    const int tid   = blockIdx.x * blockDim.x + threadIdx.x;
    const int total = (FIN / 32) * 4 * 32;
    if (tid >= total) return;
    const int lane = tid & 31;
    const int t    = (tid >> 5) & 3;
    const int s    = tid >> 7;
    const int col  = t * 16 + (lane & 15);
    const int kb   = (lane < 16) ? 0 : 8;
    _Float16* out  = frag + (size_t)tid * 16;
#pragma unroll
    for (int i = 0; i < 16; ++i)
        out[i] = (_Float16)w[(size_t)(s * 32 + kb + (i & 7) + ((i >> 3) << 4)) * HID + col];
}

// ---------------------------------------------------------------------------
// Node transform: xl = x@wl + bl, xr = x@wr + br   (WMMA f16 in, f32 acc)
// One wave computes a 16-row x 64-col tile of both outputs.
// B comes pre-fragmented (contiguous 32B per lane); A via float4 + cvt_pk.
// ---------------------------------------------------------------------------
template <int FIN>
__global__ __launch_bounds__(128)
void gat_transform(const float* __restrict__ x,
                   const _Float16* __restrict__ fl, const float* __restrict__ bl,
                   const _Float16* __restrict__ fr, const float* __restrict__ br,
                   float* __restrict__ xl, float* __restrict__ xr, int N)
{
    const int lane = threadIdx.x & 31;
    const int wave = blockIdx.x * (blockDim.x >> 5) + (threadIdx.x >> 5);
    const int row0 = wave * 16;
    if (row0 >= N) return;                 // wave-uniform: EXEC stays all-ones

    const int r   = lane & 15;
    const int hi  = lane >> 4;             // 0 | 1
    const int akb = hi ? 8 : 0;            // A k-sub-base
    int row = row0 + r; if (row >= N) row = N - 1;  // branch-free clamp

    v8f accl[4] = {};
    v8f accr[4] = {};

#pragma unroll
    for (int s = 0; s < FIN / 32; ++s) {
        // ---- A fragment: 16x32 f16, two contiguous 8-float runs per lane ----
        const float* xp = x + (long)row * FIN + s * 32;
        const float4 a0 = *(const float4*)(xp + akb);
        const float4 a1 = *(const float4*)(xp + akb + 4);
        const float4 a2 = *(const float4*)(xp + 16 + akb);
        const float4 a3 = *(const float4*)(xp + 16 + akb + 4);
        v16h a;
        a[0]  = (_Float16)a0.x; a[1]  = (_Float16)a0.y;
        a[2]  = (_Float16)a0.z; a[3]  = (_Float16)a0.w;
        a[4]  = (_Float16)a1.x; a[5]  = (_Float16)a1.y;
        a[6]  = (_Float16)a1.z; a[7]  = (_Float16)a1.w;
        a[8]  = (_Float16)a2.x; a[9]  = (_Float16)a2.y;
        a[10] = (_Float16)a2.z; a[11] = (_Float16)a2.w;
        a[12] = (_Float16)a3.x; a[13] = (_Float16)a3.y;
        a[14] = (_Float16)a3.z; a[15] = (_Float16)a3.w;

#pragma unroll
        for (int t = 0; t < 4; ++t) {
            const size_t foff = ((size_t)(s * 4 + t) * 32 + lane) * 16;
            const v16h b  = *(const v16h*)(fl + foff);
            accl[t] = __builtin_amdgcn_wmma_f32_16x16x32_f16(
                false, a, false, b, (short)0, accl[t], false, false);
            const v16h b2 = *(const v16h*)(fr + foff);
            accr[t] = __builtin_amdgcn_wmma_f32_16x16x32_f16(
                false, a, false, b2, (short)0, accr[t], false, false);
        }
    }

    // ---- store C tiles: elem vr -> M = vr + 8*hi, N = t*16 + r ----
#pragma unroll
    for (int t = 0; t < 4; ++t) {
        const int col = t * 16 + r;
#pragma unroll
        for (int vr = 0; vr < 8; ++vr) {
            const int orow = row0 + vr + (hi ? 8 : 0);
            if (orow < N) {
                xl[(long)orow * HID + col] = accl[t][vr] + bl[col];
                xr[(long)orow * HID + col] = accr[t][vr] + br[col];
            }
        }
    }
}

// ---------------------------------------------------------------------------
__global__ void gat_init(unsigned* __restrict__ amax, float* __restrict__ denom,
                         float* __restrict__ acc, int N)
{
    const int i = blockIdx.x * blockDim.x + threadIdx.x;
    if (i < N * NH) { amax[i] = 0x007FFFFFu; /* ord(-inf) */ denom[i] = 0.f; }
    if (i < N * HID) acc[i] = 0.f;
}

// ---------------------------------------------------------------------------
// Pass 1 over edges: logits a[e,h] + running segment max over dst
// ---------------------------------------------------------------------------
__global__ void gat_edge_logits(const int* __restrict__ ei, int E, int N,
                                const float* __restrict__ xl,
                                const float* __restrict__ xr,
                                const float* __restrict__ att,   // [NH,CH]
                                float* __restrict__ aout,
                                unsigned* __restrict__ amax)
{
    const long t  = (long)blockIdx.x * blockDim.x + threadIdx.x;
    const long EN = (long)E + N;
    if (t >= EN * NH) return;
    const int  h = (int)(t & 3);
    const long e = t >> 2;
    int s, d;
    if (e < E) { s = ei[e]; d = ei[(long)E + e]; } else { s = d = (int)(e - E); }

    const float4* pl = (const float4*)(xl + (long)s * HID + h * CH);
    const float4* pr = (const float4*)(xr + (long)d * HID + h * CH);
    const float4* pa = (const float4*)(att + h * CH);
    float acc = 0.f;
#pragma unroll
    for (int q = 0; q < 4; ++q) {
        const float4 vl = pl[q], vr = pr[q], va = pa[q];
        acc += va.x * lrelu(vl.x + vr.x, 0.2f);
        acc += va.y * lrelu(vl.y + vr.y, 0.2f);
        acc += va.z * lrelu(vl.z + vr.z, 0.2f);
        acc += va.w * lrelu(vl.w + vr.w, 0.2f);
    }
    aout[e * NH + h] = acc;
    atomicMax(amax + (long)d * NH + h, f2ord(acc));
}

// ---------------------------------------------------------------------------
// Pass 2: ex = exp(a - amax[dst]); denom[dst] += ex; acc[dst] += xl[src]*ex
// (normalization folded out: sum(xl*ex)/sum(ex) == sum(xl*alpha))
// ---------------------------------------------------------------------------
__global__ void gat_edge_scatter(const int* __restrict__ ei, int E, int N,
                                 const float* __restrict__ xl,
                                 const float* __restrict__ aout,
                                 const unsigned* __restrict__ amax,
                                 float* __restrict__ denom,
                                 float* __restrict__ acc)
{
    const long t  = (long)blockIdx.x * blockDim.x + threadIdx.x;
    const long EN = (long)E + N;
    if (t >= EN * NH) return;
    const int  h = (int)(t & 3);
    const long e = t >> 2;
    int s, d;
    if (e < E) { s = ei[e]; d = ei[(long)E + e]; } else { s = d = (int)(e - E); }

    const float m  = ord2f(amax[(long)d * NH + h]);
    const float ex = __expf(aout[e * NH + h] - m);
    atomicAdd(denom + (long)d * NH + h, ex);
    const float* ps = xl + (long)s * HID + h * CH;
    float*       pd = acc + (long)d * HID + h * CH;
#pragma unroll
    for (int c = 0; c < CH; ++c) atomicAdd(pd + c, ps[c] * ex);
}

// ---------------------------------------------------------------------------
__global__ void gat_finalize(float* __restrict__ acc,
                             const float* __restrict__ denom,
                             const float* __restrict__ bias, int N, int apply_act)
{
    const int i = blockIdx.x * blockDim.x + threadIdx.x;
    if (i >= N * HID) return;
    const int n = i >> 6, j = i & 63;
    float v = acc[i] / (denom[n * NH + (j >> 4)] + 1e-16f) + bias[j];
    if (apply_act) v = lrelu(v, 0.01f);
    acc[i] = v;
}

// ---------------------------------------------------------------------------
// MLP head: out = lrelu(h@fc1+b1, .01) @ fc2 + b2, fc1 staged in LDS
// ---------------------------------------------------------------------------
__global__ __launch_bounds__(256)
void gat_mlp(const float* __restrict__ h,
             const float* __restrict__ w1, const float* __restrict__ b1,
             const float* __restrict__ w2, const float* __restrict__ b2,
             float* __restrict__ out, int N)
{
    __shared__ float sw1[HID * 32];
    __shared__ float sw2[32];
    __shared__ float sb1[32];
    for (int i = threadIdx.x; i < HID * 32; i += blockDim.x) sw1[i] = w1[i];
    if (threadIdx.x < 32) { sw2[threadIdx.x] = w2[threadIdx.x]; sb1[threadIdx.x] = b1[threadIdx.x]; }
    __syncthreads();

    const int n = blockIdx.x * blockDim.x + threadIdx.x;
    if (n >= N) return;
    float hv[HID];
    const float4* hp = (const float4*)(h + (long)n * HID);
#pragma unroll
    for (int q = 0; q < 16; ++q) {
        const float4 v = hp[q];
        hv[q * 4 + 0] = v.x; hv[q * 4 + 1] = v.y;
        hv[q * 4 + 2] = v.z; hv[q * 4 + 3] = v.w;
    }
    float o = b2[0];
#pragma unroll 4
    for (int j = 0; j < 32; ++j) {
        float a = sb1[j];
#pragma unroll
        for (int k = 0; k < HID; ++k) a += hv[k] * sw1[k * 32 + j];
        o += lrelu(a, 0.01f) * sw2[j];
    }
    out[n] = o;
}

// ---------------------------------------------------------------------------
static inline size_t align256(size_t v) { return (v + 255) & ~(size_t)255; }

extern "C" void kernel_launch(void* const* d_in, const int* in_sizes, int n_in,
                              void* d_out, int out_size, void* d_ws, size_t ws_size,
                              hipStream_t stream)
{
    const float* x    = (const float*)d_in[0];
    const int*   ei   = (const int*)  d_in[1];
    const float* wl0  = (const float*)d_in[2];
    const float* bl0  = (const float*)d_in[3];
    const float* wr0  = (const float*)d_in[4];
    const float* br0  = (const float*)d_in[5];
    const float* att0 = (const float*)d_in[6];
    const float* bs0  = (const float*)d_in[7];
    const float* wl1  = (const float*)d_in[8];
    const float* bl1  = (const float*)d_in[9];
    const float* wr1  = (const float*)d_in[10];
    const float* br1  = (const float*)d_in[11];
    const float* att1 = (const float*)d_in[12];
    const float* bs1  = (const float*)d_in[13];
    const float* f1w  = (const float*)d_in[14];
    const float* f1b  = (const float*)d_in[15];
    const float* f2w  = (const float*)d_in[16];
    const float* f2b  = (const float*)d_in[17];

    const int  N  = in_sizes[0] / 128;
    const int  E  = in_sizes[1] / 2;
    const long EN = (long)E + N;

    char* base = (char*)d_ws;
    size_t off = 0;
    float*    xl    = (float*)(base + off);    off = align256(off + (size_t)N * HID * sizeof(float));
    float*    xr    = (float*)(base + off);    off = align256(off + (size_t)N * HID * sizeof(float));
    float*    hbuf  = (float*)(base + off);    off = align256(off + (size_t)N * HID * sizeof(float));
    float*    aout  = (float*)(base + off);    off = align256(off + (size_t)EN * NH * sizeof(float));
    unsigned* amax  = (unsigned*)(base + off); off = align256(off + (size_t)N * NH * sizeof(unsigned));
    float*    denom = (float*)(base + off);    off = align256(off + (size_t)N * NH * sizeof(float));
    _Float16* fl0   = (_Float16*)(base + off); off = align256(off + (size_t)128 * HID * sizeof(_Float16));
    _Float16* fr0   = (_Float16*)(base + off); off = align256(off + (size_t)128 * HID * sizeof(_Float16));
    _Float16* fl1   = (_Float16*)(base + off); off = align256(off + (size_t)64  * HID * sizeof(_Float16));
    _Float16* fr1   = (_Float16*)(base + off); off = align256(off + (size_t)64  * HID * sizeof(_Float16));

    const int waves     = (N + 15) / 16;
    const int tblocks   = (waves + 3) / 4;                 // 4 waves / block
    const unsigned eb   = (unsigned)((EN * NH + 255) / 256);
    const unsigned nb64 = (unsigned)(((long)N * HID + 255) / 256);
    const unsigned nb   = (unsigned)((N + 255) / 256);

    // ---- weight fragment prep (all up front; tiny) ----
    prep_wfrag<128><<<2, 256, 0, stream>>>(wl0, fl0);      // 512 fragments
    prep_wfrag<128><<<2, 256, 0, stream>>>(wr0, fr0);
    prep_wfrag<64> <<<1, 256, 0, stream>>>(wl1, fl1);      // 256 fragments
    prep_wfrag<64> <<<1, 256, 0, stream>>>(wr1, fr1);

    // ---- layer 0 (FIN = 128) ----
    gat_transform<128><<<tblocks, 128, 0, stream>>>(x, fl0, bl0, fr0, br0, xl, xr, N);
    gat_init<<<nb64, 256, 0, stream>>>(amax, denom, hbuf, N);
    gat_edge_logits<<<eb, 256, 0, stream>>>(ei, E, N, xl, xr, att0, aout, amax);
    gat_edge_scatter<<<eb, 256, 0, stream>>>(ei, E, N, xl, aout, amax, denom, hbuf);
    gat_finalize<<<nb64, 256, 0, stream>>>(hbuf, denom, bs0, N, 1);

    // ---- layer 1 (FIN = 64) ----
    gat_transform<64><<<tblocks, 128, 0, stream>>>(hbuf, fl1, bl1, fr1, br1, xl, xr, N);
    gat_init<<<nb64, 256, 0, stream>>>(amax, denom, hbuf, N);
    gat_edge_logits<<<eb, 256, 0, stream>>>(ei, E, N, xl, xr, att1, aout, amax);
    gat_edge_scatter<<<eb, 256, 0, stream>>>(ei, E, N, xl, aout, amax, denom, hbuf);
    gat_finalize<<<nb64, 256, 0, stream>>>(hbuf, denom, bs1, N, 0);

    // ---- MLP head ----
    gat_mlp<<<nb, 256, 0, stream>>>(hbuf, f1w, f1b, f2w, f2b, (float*)d_out, N);
}
